// KNRM_74878459838958
// MI455X (gfx1250) — matrix operand order repaired and checked by reading, not compiled, not run on hardware
//
#include <hip/hip_runtime.h>

typedef __attribute__((ext_vector_type(16))) _Float16 v16h;
typedef __attribute__((ext_vector_type(8)))  _Float16 v8h;
typedef __attribute__((ext_vector_type(8)))  float    v8f;

#define NK       11
#define QLEN     32
#define DLEN     256
#define EDIM     128
#define DCHUNK   64
#define HSTR     (EDIM + 8)      // f16 LDS row stride (bytes: 272, multiple of 16)
#define TMSTR    (DCHUNK + 4)    // f32 tm row stride (bank stagger)
#define NTHREADS 256
#define INV_SQRT_2PI 0.3989422804014327f

// mu = [1.0, 0.9, 0.7, ..., -0.9]; sigma = [1e-4, 0.1 x10]; coef = -1/(2*sigma^2)
__constant__ float c_mu[NK]   = {1.0f, 0.9f, 0.7f, 0.5f, 0.3f, 0.1f,
                                 -0.1f, -0.3f, -0.5f, -0.7f, -0.9f};
__constant__ float c_coef[NK] = {-5.0e7f, -50.f, -50.f, -50.f, -50.f, -50.f,
                                 -50.f, -50.f, -50.f, -50.f, -50.f};

// Gather NROWS embedding rows, L2-normalize, write f16 rows (stride HSTR) to LDS.
template <int NROWS, int EPR>
__device__ __forceinline__ void gather_norm(const float* __restrict__ emb,
                                            const int* __restrict__ toks,
                                            _Float16* out, float* mask,
                                            float* red, int tid) {
  constexpr int TPR = NTHREADS / NROWS;   // threads per row
  const int r  = tid / TPR;
  const int li = tid % TPR;
  const int c0 = li * EPR;
  const int tok = toks[r];
  if (li == 0) mask[r] = (tok > 0) ? 1.0f : 0.0f;
  const float* src = emb + (size_t)tok * EDIM + c0;
  float vals[EPR];
  float ss = 0.0f;
#pragma unroll
  for (int i = 0; i < EPR; i += 4) {
    float4 t = *(const float4*)(src + i);
    vals[i] = t.x; vals[i + 1] = t.y; vals[i + 2] = t.z; vals[i + 3] = t.w;
    ss += t.x * t.x + t.y * t.y + t.z * t.z + t.w * t.w;
  }
  red[tid] = ss;
  __syncthreads();
  float tot = 0.0f;
#pragma unroll
  for (int i = 0; i < TPR; ++i) tot += red[r * TPR + i];
  const float rn = rsqrtf(tot);
  _Float16* dst = out + r * HSTR + c0;
#pragma unroll
  for (int j = 0; j < EPR / 8; ++j) {
    v8h h;
#pragma unroll
    for (int i = 0; i < 8; ++i) h[i] = (_Float16)(vals[j * 8 + i] * rn);
    *(v8h*)(dst + j * 8) = h;
  }
}

__global__ __launch_bounds__(NTHREADS) void knrm_kernel(
    const int* __restrict__ qtok, const int* __restrict__ dtok,
    const float* __restrict__ emb, const float* __restrict__ fcw,
    float* __restrict__ out) {
  __shared__ __align__(16) _Float16 qs[QLEN * HSTR];
  __shared__ __align__(16) _Float16 dsm[DCHUNK * HSTR];
  __shared__ float tmS[QLEN * TMSTR];
  __shared__ float accum[QLEN * NK];
  __shared__ float red[NTHREADS];
  __shared__ float qmask[QLEN];
  __shared__ float dmask[DCHUNK];

  const int b   = blockIdx.x;
  const int tid = threadIdx.x;

  for (int i = tid; i < QLEN * NK; i += NTHREADS) accum[i] = 0.0f;

  // ---- queries: 32 rows, 8 threads/row, 16 elems/thread ----
  gather_norm<QLEN, 16>(emb, qtok + b * QLEN, qs, qmask, red, tid);

  const int wave = tid >> 5;            // 0..7
  const int lane = tid & 31;
  const int mt   = wave >> 2;           // 0..1  (query tile)
  const int nt   = wave & 3;            // 0..3  (doc tile within chunk)
  const int am   = mt * 16 + (lane & 15);          // A-matrix row
  const int bn   = nt * 16 + (lane & 15);          // B-matrix col (doc row)
  const int abase = (lane >> 4) * 8;               // A K-offset per ISA layout
  const int bbase = (lane >> 4) * 8;               // B K-offset per ISA layout

  for (int ch = 0; ch < 4; ++ch) {
    __syncthreads();  // prior readers of red/dmask/dsm/tmS done; qs visible

    // ---- docs: 64 rows, 4 threads/row, 32 elems/thread ----
    gather_norm<DCHUNK, 32>(emb, dtok + b * DLEN + ch * DCHUNK, dsm, dmask,
                            red, tid);
    __syncthreads();

    // ---- WMMA: each wave owns one 16x16 tile, K=128 in 4 steps ----
    {
      v8f c = {};
#pragma unroll
      for (int kk = 0; kk < EDIM; kk += 32) {
        union { v8h h8[2]; v16h h16; } A, B;
        A.h8[0] = *(const v8h*)&qs[am * HSTR + kk + abase];
        A.h8[1] = *(const v8h*)&qs[am * HSTR + kk + 16 + abase];
        B.h8[0] = *(const v8h*)&dsm[bn * HSTR + kk + bbase];
        B.h8[1] = *(const v8h*)&dsm[bn * HSTR + kk + 16 + bbase];
        c = __builtin_amdgcn_wmma_f32_16x16x32_f16(false, A.h16, false, B.h16,
                                                   (short)0, c, false, false);
      }
      // scatter C fragment: lane holds col n=lane%16, rows v + (lane/16)*8
      const int row0 = mt * 16 + (lane >> 4) * 8;
      const int col  = nt * 16 + (lane & 15);
#pragma unroll
      for (int v = 0; v < 8; ++v)
        tmS[(row0 + v) * TMSTR + col] = c[v];
    }
    __syncthreads();

    // ---- kernel pooling over this doc chunk (no atomics) ----
    for (int p = tid; p < QLEN * NK; p += NTHREADS) {
      const int r = p / NK;
      const int k = p % NK;
      const float mu = c_mu[k], coef = c_coef[k];
      const float* row = &tmS[r * TMSTR];
      float s = 0.0f;
#pragma unroll 8
      for (int d = 0; d < DCHUNK; ++d) {
        const float diff = row[d] - mu;
        s += dmask[d] * __expf(coef * diff * diff);
      }
      accum[p] += s;
    }
  }
  __syncthreads();

  // ---- log pooling + fc weight, block reduction ----
  float part = 0.0f;
  for (int p = tid; p < QLEN * NK; p += NTHREADS) {
    const int r = p / NK;
    const int k = p % NK;
    float v = accum[p] * INV_SQRT_2PI;
    v = fmaxf(v, 1e-10f);
    part += qmask[r] * __logf(v) * fcw[k];
  }
  red[tid] = part;
  __syncthreads();
  for (int s = NTHREADS / 2; s > 0; s >>= 1) {
    if (tid < s) red[tid] += red[tid + s];
    __syncthreads();
  }
  if (tid == 0) out[b] = red[0];
}

extern "C" void kernel_launch(void* const* d_in, const int* in_sizes, int n_in,
                              void* d_out, int out_size, void* d_ws,
                              size_t ws_size, hipStream_t stream) {
  (void)n_in; (void)out_size; (void)d_ws; (void)ws_size;
  const int*   qtok = (const int*)d_in[0];    // [B, 32]
  const int*   dtok = (const int*)d_in[1];    // [B, 256]
  const float* emb  = (const float*)d_in[2];  // [32000, 128]
  const float* fcw  = (const float*)d_in[3];  // [1, 11]
  float*       out  = (float*)d_out;          // [B]
  const int B = in_sizes[0] / QLEN;           // 512
  knrm_kernel<<<B, NTHREADS, 0, stream>>>(qtok, dtok, emb, fcw, out);
}
